// RandomProjectionQuantizer_19980187861914
// MI455X (gfx1250) — matrix operand order repaired and checked by reading, not compiled, not run on hardware
//
#include <hip/hip_runtime.h>
#include <hip/hip_bf16.h>

// Problem constants (match reference)
#define B_ 16
#define L_ 4096
#define D_ 512
#define Q_ 256
#define C_ 1024

#define MTILE 32          // rows per workgroup
#define PX 520            // LDS pitch for X slab (bf16), padded
#define PT 264            // LDS pitch for T slab (bf16), padded
#define PS 132            // LDS pitch for score chunk (f32), padded
#define CCHUNK 128        // codebook columns per chunk

typedef __attribute__((ext_vector_type(16))) __bf16        v16bf;
typedef __attribute__((ext_vector_type(8)))  float         v8f;
typedef __attribute__((ext_vector_type(8)))  unsigned int  v8u;

static __device__ __forceinline__ unsigned short f2bf(float f) {
  // round-to-nearest-even f32 -> bf16
  unsigned int x = __float_as_uint(f);
  unsigned int r = x + 0x7FFFu + ((x >> 16) & 1u);
  return (unsigned short)(r >> 16);
}

// ---------------- prep kernels ----------------

__global__ __launch_bounds__(256) void rpq_prep_proj(const float* __restrict__ w,
                                                     unsigned short* __restrict__ wbf) {
  int i = blockIdx.x * 256 + threadIdx.x;
  if (i < Q_ * D_) wbf[i] = f2bf(w[i]);
}

// codebook (Q,C) f32 -> cbT (C,Q) bf16  +  c2[c] = sum_q cb[q][c]^2 (f32)
__global__ __launch_bounds__(256) void rpq_prep_cb(const float* __restrict__ cb,
                                                   unsigned short* __restrict__ cbT,
                                                   float* __restrict__ c2) {
  int c = blockIdx.x * 256 + threadIdx.x;
  if (c >= C_) return;
  float s = 0.f;
  for (int q = 0; q < Q_; ++q) {
    float v = cb[q * C_ + c];          // coalesced across threads
    s += v * v;
    cbT[(size_t)c * Q_ + q] = f2bf(v);
  }
  c2[c] = s;
}

// ---------------- main fused kernel ----------------
// grid = (B*L)/MTILE blocks of 256 threads (8 wave32)

__global__ __launch_bounds__(256) void rpq_main(const float* __restrict__ X,
                                                const unsigned short* __restrict__ Wbf,
                                                const unsigned short* __restrict__ cbT,
                                                const float* __restrict__ c2,
                                                float* __restrict__ out) {
  // region A: X slab (bf16) during GEMM1, reused as score chunk (f32) afterwards
  __shared__ __align__(16) unsigned char ldsA[MTILE * PX * 2];   // 33280 B
  __shared__ __align__(16) unsigned short Tb[MTILE * PT];        // 16896 B
  __shared__ float redV[256];
  __shared__ int   redI[256];

  unsigned short* Xb = (unsigned short*)ldsA;
  float*          Sc = (float*)ldsA;

  const int tid  = threadIdx.x;
  const int lane = tid & 31;
  const int wv   = tid >> 5;        // wave id 0..7
  const int half = lane >> 4;       // 0: lanes 0-15, 1: lanes 16-31
  const int l16  = lane & 15;
  const int rowbase = blockIdx.x * MTILE;

  // ---- cooperative load: X slab f32 -> bf16 in LDS ----
  {
    const float4* Xg = (const float4*)(X + (size_t)rowbase * D_);
    #pragma unroll
    for (int i = 0; i < 16; ++i) {
      int flat4 = i * 256 + tid;            // 4096 float4 per slab
      int r  = flat4 >> 7;                  // 128 float4 per row
      int c0 = (flat4 & 127) << 2;
      float4 v = Xg[flat4];
      unsigned int lo = (unsigned)f2bf(v.x) | ((unsigned)f2bf(v.y) << 16);
      unsigned int hi = (unsigned)f2bf(v.z) | ((unsigned)f2bf(v.w) << 16);
      unsigned int* p = (unsigned int*)&Xb[r * PX + c0];
      p[0] = lo; p[1] = hi;
    }
  }
  __syncthreads();

  // ---- GEMM1: T[32x256] = Xbf[32x512] @ W^T, bf16 WMMA, f32 accum ----
  {
    const int mt  = wv & 1;          // this wave's M tile (rows mt*16..)
    const int qt0 = wv >> 1;         // N tiles qt0 + 4*j, j=0..3
    v8f acc[4] = {};
    const int arow = mt * 16 + l16;
    #pragma unroll 2
    for (int kk = 0; kk < 16; ++kk) {
      const int kb = kk * 32;
      v8u au;
      #pragma unroll
      for (int i = 0; i < 8; ++i) {
        // A 16x32 bf16 layout: lanes0-15 K{0..7,16..23}, lanes16-31 K{8..15,24..31}
        int ka = kb + ((i < 4) ? (2 * i + 8 * half) : (16 + 2 * (i - 4) + 8 * half));
        au[i] = *(const unsigned int*)&Xb[arow * PX + ka];
      }
      v16bf a = __builtin_bit_cast(v16bf, au);
      #pragma unroll
      for (int j = 0; j < 4; ++j) {
        const int qcol = (qt0 + 4 * j) * 16 + l16;
        v8u bu;
        #pragma unroll
        for (int i = 0; i < 8; ++i) {
          // B 32x16 bf16 layout: lanes0-15 K{0..7,16..23}, lanes16-31 K{8..15,24..31}
          int kbv = kb + ((i < 4) ? (2 * i + 8 * half) : (16 + 2 * (i - 4) + 8 * half));
          bu[i] = *(const unsigned int*)&Wbf[qcol * D_ + kbv];   // B(k,q)=W[q][k]
        }
        v16bf b = __builtin_bit_cast(v16bf, bu);
        acc[j] = __builtin_amdgcn_wmma_f32_16x16x32_bf16(
            false, a, false, b, (short)0, acc[j], false, false);
      }
    }
    // C/D layout: VGPR r holds (m = r + 8*half, n = l16)
    #pragma unroll
    for (int j = 0; j < 4; ++j)
      #pragma unroll
      for (int r = 0; r < 8; ++r) {
        int m = mt * 16 + r + 8 * half;
        Tb[m * PT + (qt0 + 4 * j) * 16 + l16] = f2bf(acc[j][r]);
      }
  }
  __syncthreads();   // Tb ready; Xb dead -> Sc may reuse region A

  // ---- GEMM2 + fused score + running argmin, C in chunks of 128 ----
  const int srow = tid >> 3;         // 0..31 : row scanned by this thread
  const int part = tid & 7;          // 8 threads share a row
  float bv = 3.4e38f;
  int   bi = 0;

  for (int ch = 0; ch < C_ / CCHUNK; ++ch) {
    const int cbase = ch * CCHUNK;
    {
      const int mt  = wv & 1;
      const int ct0 = wv >> 1;       // ct0, ct0+4
      v8f acc[2] = {};
      const int arow = mt * 16 + l16;
      #pragma unroll
      for (int kk = 0; kk < 8; ++kk) {
        const int kb = kk * 32;
        v8u au;
        #pragma unroll
        for (int i = 0; i < 8; ++i) {
          int ka = kb + ((i < 4) ? (2 * i + 8 * half) : (16 + 2 * (i - 4) + 8 * half));
          au[i] = *(const unsigned int*)&Tb[arow * PT + ka];
        }
        v16bf a = __builtin_bit_cast(v16bf, au);
        #pragma unroll
        for (int j = 0; j < 2; ++j) {
          const int ccol = cbase + (ct0 + 4 * j) * 16 + l16;
          v8u bu;
          #pragma unroll
          for (int i = 0; i < 8; ++i) {
            int kbv = kb + ((i < 4) ? (2 * i + 8 * half) : (16 + 2 * (i - 4) + 8 * half));
            bu[i] = *(const unsigned int*)&cbT[(size_t)ccol * Q_ + kbv]; // B(k,c)=cbT[c][k]
          }
          v16bf b = __builtin_bit_cast(v16bf, bu);
          acc[j] = __builtin_amdgcn_wmma_f32_16x16x32_bf16(
              false, a, false, b, (short)0, acc[j], false, false);
        }
      }
      // score = ||c||^2 - 2 * t.c   (t2 dropped: constant per row for argmin)
      #pragma unroll
      for (int j = 0; j < 2; ++j) {
        const int ccol = cbase + (ct0 + 4 * j) * 16 + l16;
        const float c2v = c2[ccol];
        #pragma unroll
        for (int r = 0; r < 8; ++r) {
          int m = mt * 16 + r + 8 * half;
          Sc[m * PS + (ct0 + 4 * j) * 16 + l16] = c2v - 2.0f * acc[j][r];
        }
      }
    }
    __syncthreads();
    // scan this chunk: 8 threads per row, 16 cols each, lexicographic tie-break
    #pragma unroll
    for (int jj = 0; jj < 16; ++jj) {
      int col  = part * 16 + jj;
      float v  = Sc[srow * PS + col];
      int gcol = cbase + col;
      if (v < bv || (v == bv && gcol < bi)) { bv = v; bi = gcol; }
    }
    __syncthreads();   // protect Sc before next chunk's writes
  }

  // ---- combine 8 partial argmins per row, emit label ----
  redV[tid] = bv; redI[tid] = bi;
  __syncthreads();
  if (part == 0) {
    float fv = redV[tid]; int fi = redI[tid];
    #pragma unroll
    for (int p = 1; p < 8; ++p) {
      float v = redV[tid + p]; int i2 = redI[tid + p];
      if (v < fv || (v == fv && i2 < fi)) { fv = v; fi = i2; }
    }
    out[rowbase + srow] = (float)fi;   // labels as f32 (harness output dtype)
  }
}

// ---------------- launch ----------------

extern "C" void kernel_launch(void* const* d_in, const int* in_sizes, int n_in,
                              void* d_out, int out_size, void* d_ws, size_t ws_size,
                              hipStream_t stream) {
  const float* X  = (const float*)d_in[0];   // (B,L,D) f32
  const float* W  = (const float*)d_in[1];   // (Q,D)   f32
  const float* CB = (const float*)d_in[2];   // (Q,C)   f32
  (void)in_sizes; (void)n_in; (void)out_size; (void)ws_size;   // raw_signal == 1 branch

  unsigned char* ws = (unsigned char*)d_ws;
  float*          c2  = (float*)ws;                                  //   4 KB
  unsigned short* wbf = (unsigned short*)(ws + 4096);                // 256 KB
  unsigned short* cbT = (unsigned short*)(ws + 4096 + Q_ * D_ * 2);  // 512 KB

  rpq_prep_proj<<<(Q_ * D_ + 255) / 256, 256, 0, stream>>>(W, wbf);
  rpq_prep_cb<<<(C_ + 255) / 256, 256, 0, stream>>>(CB, cbT, c2);
  rpq_main<<<(B_ * L_) / MTILE, 256, 0, stream>>>(X, wbf, cbT, c2, (float*)d_out);
}
